// Encoder_58763742544532
// MI455X (gfx1250) — compile-verified, matrix operand order checked by reading, not run on hardware
//
#include <hip/hip_runtime.h>
#include <hip/hip_bf16.h>

// Problem sizes (fixed by the reference)
#define B_  1024
#define D_  784
#define H_  512
#define K_  64
#define S_  4
#define ND_ (K_ * D_)   // 50176 columns of out_w per stage
#define DMAIN_ 768      // 24 full K-chunks of 32; tail = 16

typedef __attribute__((ext_vector_type(16))) __bf16 v16bf;
typedef __attribute__((ext_vector_type(2)))  __bf16 v2bf;
typedef __attribute__((ext_vector_type(8)))  float  v8f;

union Frag { v16bf v; unsigned u[8]; };

// native RNE f32 -> bf16 (lowers to v_cvt_*bf16_f32 on gfx1250)
__device__ __forceinline__ unsigned short f2bf(float f) {
    union { __bf16 b; unsigned short u; } c;
    c.b = (__bf16)f;
    return c.u;
}
// packed pair convert: one v_cvt_pk_bf16_f32
__device__ __forceinline__ unsigned pack2(float a, float b) {
    union { v2bf v; unsigned u; } c;
    v2bf t = { (__bf16)a, (__bf16)b };
    c.v = t;
    return c.u;
}
__device__ __forceinline__ float bf2f(unsigned short h) {
    return __uint_as_float(((unsigned)h) << 16);
}

// Load one bf16 WMMA fragment (16x32 A or 32x16 B) from LDS staged as
// [16 rows/cols][32 K] ushort with row stride 34 (bank-conflict pad).
// `sm` points at the wave's 16-row subtile base.
// ISA 7.12.2 layout: lane l -> row/col = l&15, K-halfblock = 8*(l>>4);
// vector elem pairs are K-contiguous -> ds_load_b32 (pairs merge to 2addr).
__device__ __forceinline__ void load_frag(Frag& f, const unsigned short* sm,
                                          int lane) {
    const int kb = (lane >> 4) << 3;
    const int rc = lane & 15;
#pragma unroll
    for (int p = 0; p < 8; ++p) {
        int kof = ((p & 4) ? 16 : 0) + kb + ((p & 3) << 1);
        f.u[p] = *(const unsigned*)&sm[rc * 34 + kof];
    }
}

__global__ void zero_kernel(float* p, int n) {
    int i = blockIdx.x * 256 + threadIdx.x;
    if (i < n) p[i] = 0.f;
}

// ---------------------------------------------------------------------------
// Stage base trunk: h = relu(cur @ base_w + base_b), [1024,784]x[784,512],
// output stored bf16 row-major (feeds the score GEMM's A operand).
// Block = 8 waves as 4Mx2N -> 64x32 tile; K chunked by 32; 24 guard-free
// chunks + one peeled 16-wide tail (zero-padded in LDS).
// ---------------------------------------------------------------------------
__global__ __launch_bounds__(256)
void base_gemm_kernel(const float* __restrict__ cur,
                      const float* __restrict__ bw,   // [784,512]
                      const float* __restrict__ bb,   // [512]
                      unsigned short* __restrict__ hout) // [1024,512] bf16
{
    __shared__ unsigned short As[64 * 34];
    __shared__ unsigned short Bs[32 * 34];
    const int tid  = threadIdx.x;
    const int lane = tid & 31;
    const int wave = tid >> 5;
    const int wm = wave & 3, wn = wave >> 2;
    const int mblk = blockIdx.x * 64;
    const int nblk = blockIdx.y * 32;

    // staging coordinates (uniform per thread)
    const int am = tid >> 2;            // A row 0..63
    const int ac = (tid & 3) << 3;      // A col base 0,8,16,24
    const int bk = tid >> 3;            // B k-row 0..31
    const int bn = (tid & 7) << 2;      // B n base 0,4,...,28

    v8f acc;
#pragma unroll
    for (int i = 0; i < 8; ++i) acc[i] = 0.f;

    // ---- 24 guard-free chunks ----
    for (int kk = 0; kk < DMAIN_; kk += 32) {
        {   // A: cur rows [mblk,+64), cols [kk,+32), f32 -> bf16, b128 loads
            const float4* src =
                (const float4*)(cur + (size_t)(mblk + am) * D_ + kk + ac);
            float4 v0 = src[0], v1 = src[1];
            unsigned* dst = (unsigned*)&As[am * 34 + ac];
            dst[0] = pack2(v0.x, v0.y);
            dst[1] = pack2(v0.z, v0.w);
            dst[2] = pack2(v1.x, v1.y);
            dst[3] = pack2(v1.z, v1.w);
        }
        {   // B transposed: Bs[n][k] <- bw[kk+bk][nblk+bn..+3]
            float4 v = *(const float4*)(bw + (size_t)(kk + bk) * H_ + nblk + bn);
            unsigned p01 = pack2(v.x, v.y);
            unsigned p23 = pack2(v.z, v.w);
            Bs[(bn + 0) * 34 + bk] = (unsigned short)p01;
            Bs[(bn + 1) * 34 + bk] = (unsigned short)(p01 >> 16);
            Bs[(bn + 2) * 34 + bk] = (unsigned short)p23;
            Bs[(bn + 3) * 34 + bk] = (unsigned short)(p23 >> 16);
        }
        __syncthreads();
        Frag a, b;
        load_frag(a, &As[wm * 16 * 34], lane);
        load_frag(b, &Bs[wn * 16 * 34], lane);
        acc = __builtin_amdgcn_wmma_f32_16x16x32_bf16(
                  false, a.v, false, b.v, (short)0, acc, false, false);
        __syncthreads();
    }

    // ---- peeled tail: kk = 768, only 16 K valid; per-vector uniform guards
    {
        {
            unsigned* dst = (unsigned*)&As[am * 34 + ac];
            if (ac < 16) {
                const float4* src =
                    (const float4*)(cur + (size_t)(mblk + am) * D_ + DMAIN_ + ac);
                float4 v0 = src[0], v1 = src[1];
                dst[0] = pack2(v0.x, v0.y);
                dst[1] = pack2(v0.z, v0.w);
                dst[2] = pack2(v1.x, v1.y);
                dst[3] = pack2(v1.z, v1.w);
            } else {
                dst[0] = 0; dst[1] = 0; dst[2] = 0; dst[3] = 0;
            }
        }
        {
            float4 v = {0.f, 0.f, 0.f, 0.f};
            if (bk < 16)
                v = *(const float4*)(bw + (size_t)(DMAIN_ + bk) * H_ + nblk + bn);
            unsigned p01 = pack2(v.x, v.y);
            unsigned p23 = pack2(v.z, v.w);
            Bs[(bn + 0) * 34 + bk] = (unsigned short)p01;
            Bs[(bn + 1) * 34 + bk] = (unsigned short)(p01 >> 16);
            Bs[(bn + 2) * 34 + bk] = (unsigned short)p23;
            Bs[(bn + 3) * 34 + bk] = (unsigned short)(p23 >> 16);
        }
        __syncthreads();
        Frag a, b;
        load_frag(a, &As[wm * 16 * 34], lane);
        load_frag(b, &Bs[wn * 16 * 34], lane);
        acc = __builtin_amdgcn_wmma_f32_16x16x32_bf16(
                  false, a.v, false, b.v, (short)0, acc, false, false);
        __syncthreads();
    }

    // epilogue: +bias, ReLU, store bf16.  C layout: VGPR r, lane l ->
    // m = r + 8*(l>>4), n = l&15.
    const int col = nblk + wn * 16 + (lane & 15);
    const float bias = bb[col];
#pragma unroll
    for (int r = 0; r < 8; ++r) {
        int row = mblk + wm * 16 + r + ((lane >> 4) << 3);
        float v = acc[r] + bias;
        v = v > 0.f ? v : 0.f;
        hout[(size_t)row * H_ + col] = f2bf(v);
    }
}

// ---------------------------------------------------------------------------
// Score GEMM with fused MSE epilogue:
//   losses[b,k] = sum_d ( (h[b,:]@w[:,k*784+d]) + ob + sb + cur - x )^2
// Block: m-block of 64 rows x one option k; loops N over 784 (25 tiles of 32,
// last half-masked), depth H=512 (guard-free).  Never materializes the
// [B,K,D] candidate tensor.
// ---------------------------------------------------------------------------
__global__ __launch_bounds__(256)
void score_kernel(const unsigned short* __restrict__ hmat, // [1024,512] bf16
                  const float* __restrict__ w,    // [512, 50176]
                  const float* __restrict__ ob,   // [50176]
                  const float* __restrict__ sb,   // [64*784]
                  const float* __restrict__ cur,  // [1024,784]
                  const float* __restrict__ x,    // [1024,784]
                  float* __restrict__ losses)     // [1024,64]
{
    __shared__ unsigned short As[64 * 34];
    __shared__ unsigned short Bs[32 * 34];
    __shared__ float Ls[2][64];
    const int tid  = threadIdx.x;
    const int lane = tid & 31;
    const int wave = tid >> 5;
    const int wm = wave & 3, wn = wave >> 2;
    const int mblk = blockIdx.x * 64;
    const int kopt = blockIdx.y;
    const size_t wbase = (size_t)kopt * D_;

    const int am = tid >> 2;            // A row 0..63
    const int ac = (tid & 3) << 3;      // A col base 0,8,16,24
    const int bk = tid >> 3;            // B k-row 0..31
    const int bn = (tid & 7) << 2;      // B n base 0,4,...,28

    float lacc[8];
#pragma unroll
    for (int r = 0; r < 8; ++r) lacc[r] = 0.f;

    for (int n0 = 0; n0 < D_; n0 += 32) {
        const bool nvalid = (n0 + bn) < D_;   // uniform per thread, per n-tile
        v8f acc;
#pragma unroll
        for (int i = 0; i < 8; ++i) acc[i] = 0.f;

        for (int kk = 0; kk < H_; kk += 32) {
            {   // stage A: h rows [mblk,+64), cols [kk,+32) — already bf16
                const uint4* src =
                    (const uint4*)(hmat + (size_t)(mblk + am) * H_ + kk + ac);
                uint4 v = *src;
                unsigned* dst = (unsigned*)&As[am * 34 + ac];
                dst[0] = v.x; dst[1] = v.y; dst[2] = v.z; dst[3] = v.w;
            }
            {   // stage B transposed: w rows [kk,+32), cols wbase+[n0,+32)
                if (kk + 32 < H_)  // warm L2/L0 for the next depth chunk
                    __builtin_prefetch(
                        w + (size_t)(kk + 32 + bk) * ND_ + wbase + n0, 0, 1);
                float4 v = {0.f, 0.f, 0.f, 0.f};
                if (nvalid)
                    v = *(const float4*)(w + (size_t)(kk + bk) * ND_ + wbase
                                           + n0 + bn);
                unsigned p01 = pack2(v.x, v.y);
                unsigned p23 = pack2(v.z, v.w);
                Bs[(bn + 0) * 34 + bk] = (unsigned short)p01;
                Bs[(bn + 1) * 34 + bk] = (unsigned short)(p01 >> 16);
                Bs[(bn + 2) * 34 + bk] = (unsigned short)p23;
                Bs[(bn + 3) * 34 + bk] = (unsigned short)(p23 >> 16);
            }
            __syncthreads();
            Frag a, b;
            load_frag(a, &As[wm * 16 * 34], lane);
            load_frag(b, &Bs[wn * 16 * 34], lane);
            acc = __builtin_amdgcn_wmma_f32_16x16x32_bf16(
                      false, a.v, false, b.v, (short)0, acc, false, false);
            __syncthreads();
        }

        // fused epilogue: cand - x, square, accumulate per row (mask N tail)
        int ncol = n0 + wn * 16 + (lane & 15);
        if (ncol < D_) {
            float addv = ob[wbase + ncol] + sb[wbase + ncol];
#pragma unroll
            for (int r = 0; r < 8; ++r) {
                int row = mblk + wm * 16 + r + ((lane >> 4) << 3);
                float t = acc[r] + addv + cur[(size_t)row * D_ + ncol]
                                 - x[(size_t)row * D_ + ncol];
                lacc[r] = fmaf(t, t, lacc[r]);
            }
        }
    }

    // reduce the 16 N-lanes of each half-wave, then combine the two N-halves
#pragma unroll
    for (int r = 0; r < 8; ++r) {
        float v = lacc[r];
        v += __shfl_xor(v, 1, 32);
        v += __shfl_xor(v, 2, 32);
        v += __shfl_xor(v, 4, 32);
        v += __shfl_xor(v, 8, 32);
        if ((lane & 15) == 0)
            Ls[wn][wm * 16 + r + ((lane >> 4) << 3)] = v;
    }
    __syncthreads();
    if (tid < 64)
        losses[(size_t)(mblk + tid) * K_ + kopt] = Ls[0][tid] + Ls[1][tid];
}

// argmin over K=64 options per row (first-min tie rule like jnp.argmin)
__global__ void argmin_kernel(const float* __restrict__ losses,
                              int* __restrict__ enc, int s) {
    int b = blockIdx.x * 256 + threadIdx.x;
    if (b >= B_) return;
    const float* L = losses + (size_t)b * K_;
    float best = L[0];
    int bi = 0;
#pragma unroll 4
    for (int k = 1; k < K_; ++k) {
        float v = L[k];
        if (v < best) { best = v; bi = k; }
    }
    enc[b * S_ + s] = bi;
}

// Recompute only the winning candidate per row (gathered GEMV, L2-resident
// weights) and update cur in place; last stage also writes recon to d_out.
__global__ __launch_bounds__(256)
void select_kernel(const unsigned short* __restrict__ hmat,
                   const float* __restrict__ w,
                   const float* __restrict__ ob,
                   const float* __restrict__ sb,
                   const int* __restrict__ enc, int s,
                   float* __restrict__ cur,
                   float* __restrict__ recon) {
    __shared__ float hs[H_];
    const int b = blockIdx.x;
    const int tid = threadIdx.x;
    for (int i = tid; i < H_; i += 256) hs[i] = bf2f(hmat[(size_t)b * H_ + i]);
    __syncthreads();
    const int idx = enc[b * S_ + s];
    const size_t base = (size_t)idx * D_;
    for (int d = tid; d < D_; d += 256) {
        const float* wp = w + base + d;
        float sum = 0.f;
#pragma unroll 8
        for (int hh = 0; hh < H_; ++hh)
            sum = fmaf(hs[hh], wp[(size_t)hh * ND_], sum);
        float nc = cur[(size_t)b * D_ + d] + sum + ob[base + d] + sb[base + d];
        cur[(size_t)b * D_ + d] = nc;
        if (recon) recon[(size_t)b * D_ + d] = nc;
    }
}

extern "C" void kernel_launch(void* const* d_in, const int* in_sizes, int n_in,
                              void* d_out, int out_size, void* d_ws, size_t ws_size,
                              hipStream_t stream) {
    const float* x  = (const float*)d_in[0];  // inputs     [B,D]
    const float* bw = (const float*)d_in[1];  // base_w     [D,H]
    const float* bb = (const float*)d_in[2];  // base_b     [H]
    const float* ow = (const float*)d_in[3];  // out_w      [S,H,K*D]
    const float* ob = (const float*)d_in[4];  // out_b      [S,K*D]
    const float* sb = (const float*)d_in[5];  // stage_bias [S,K,D]

    // workspace: cur f32 | h bf16 | losses f32   (~4.5 MB)
    char* ws = (char*)d_ws;
    float* cur = (float*)ws;
    size_t off = (((size_t)B_ * D_ * 4) + 255) & ~(size_t)255;
    unsigned short* h = (unsigned short*)(ws + off);
    off += (((size_t)B_ * H_ * 2) + 255) & ~(size_t)255;
    float* losses = (float*)(ws + off);

    // outputs: encodings [B,S] int32 (bit pattern) then recon [B,D] f32
    int*   enc   = (int*)d_out;
    float* recon = (float*)d_out + (size_t)B_ * S_;

    zero_kernel<<<(B_ * D_ + 255) / 256, 256, 0, stream>>>(cur, B_ * D_);

    for (int s = 0; s < S_; ++s) {
        const float* sw  = ow + (size_t)s * H_ * ND_;
        const float* sob = ob + (size_t)s * ND_;
        const float* ssb = sb + (size_t)s * ND_;
        base_gemm_kernel<<<dim3(B_ / 64, H_ / 32), 256, 0, stream>>>(cur, bw, bb, h);
        score_kernel<<<dim3(B_ / 64, K_), 256, 0, stream>>>(h, sw, sob, ssb,
                                                            cur, x, losses);
        argmin_kernel<<<(B_ + 255) / 256, 256, 0, stream>>>(losses, enc, s);
        select_kernel<<<B_, 256, 0, stream>>>(h, sw, sob, ssb, enc, s, cur,
                                              (s == S_ - 1) ? recon : nullptr);
    }
}
